// Attention_S_23785528885553
// MI455X (gfx1250) — matrix operand order repaired and checked by reading, not compiled, not run on hardware
//
#include <hip/hip_runtime.h>

// ---------------------------------------------------------------------------
// Attention for x[B=2, C=256, H=64, W=64], 8 heads, d=32.
// bf16 WMMA (v_wmma_f32_16x16x32_bf16) everywhere, f32 accumulate.
// Flash attention computes S^T = K*Q^T so softmax rows live in-lane.
// ---------------------------------------------------------------------------

typedef __attribute__((ext_vector_type(16))) __bf16 v16bf;
typedef __attribute__((ext_vector_type(2)))  __bf16 v2bf;
typedef __attribute__((ext_vector_type(8)))  float  v8f;

union Frag16 {
    v16bf v;
    unsigned short u[16];
    uint4 q[2];
};

__device__ __forceinline__ unsigned short bfbits(float f) {
    union { __bf16 h; unsigned short u; } c; c.h = (__bf16)f; return c.u;
}
__device__ __forceinline__ unsigned packbf(float a, float b) {
    union { v2bf v; unsigned u; } c;
    c.v[0] = (__bf16)a; c.v[1] = (__bf16)b;
    return c.u;
}

__device__ __forceinline__ v8f wmma_bf16(v16bf a, v16bf b, v8f c) {
    return __builtin_amdgcn_wmma_f32_16x16x32_bf16(false, a, false, b,
                                                   (short)0, c, false, false);
}

#define NTOK   4096      // H*W
#define NROWS  8192      // B*N
#define CDIM   256
#define HEADS  8
#define DHEAD  32
#define LSTR   40        // padded LDS row stride (ushorts) for bf16 tiles

// ---------------------------------------------------------------------------
// Kernel 1: QKV projection. tokens[bn][c] = x[b][c][n]; qkv = tokens@W + b.
// Q,K -> bf16 [row][256]; V -> bf16 transposed [b][h][d][N].
// Block 128 (4 waves); tile: 64 rows x 64 cols; K-step 32.
// LDS tiles staged in bf16, fragment-major: A=[row][k], B=[col][k].
// ---------------------------------------------------------------------------
__global__ __launch_bounds__(128) void qkv_kernel(
    const float* __restrict__ x, const float* __restrict__ w,
    const float* __restrict__ bias,
    unsigned short* __restrict__ qb, unsigned short* __restrict__ kb,
    unsigned short* __restrict__ vtb)
{
    __shared__ unsigned short Asb[64 * LSTR];   // [row][k] bf16
    __shared__ unsigned short Bsb[64 * LSTR];   // [col][k] bf16
    const int tid  = threadIdx.x;
    const int lane = tid & 31, wave = tid >> 5;
    const int l15  = lane & 15, half = lane >> 4;
    const int rowBase = blockIdx.x * 64;     // token rows [0, 8192)
    const int colBase = blockIdx.y * 64;     // qkv cols [0, 768)
    const int gb = rowBase >> 12;            // batch (tile never crosses)
    const int gn0 = rowBase & 4095;

    v8f acc[4];
    #pragma unroll
    for (int t = 0; t < 4; ++t) acc[t] = v8f{};

    const int rloc = wave * 16 + l15;

    for (int k0 = 0; k0 < CDIM; k0 += 32) {
        // stage both tiles: coalesced float4 global loads, cvt, bf16 LDS
        #pragma unroll
        for (int i = 0; i < 4; ++i) {
            int q  = i * 128 + tid;          // 0..511
            int kk = q >> 4;                 // 0..31
            int n4 = (q & 15) * 4;           // 0..60
            float4 xv = *(const float4*)&x[(size_t)(gb * CDIM + k0 + kk) * NTOK + gn0 + n4];
            Asb[(n4 + 0) * LSTR + kk] = bfbits(xv.x);
            Asb[(n4 + 1) * LSTR + kk] = bfbits(xv.y);
            Asb[(n4 + 2) * LSTR + kk] = bfbits(xv.z);
            Asb[(n4 + 3) * LSTR + kk] = bfbits(xv.w);
            float4 wv = *(const float4*)&w[(k0 + kk) * 768 + colBase + n4];
            Bsb[(n4 + 0) * LSTR + kk] = bfbits(wv.x);
            Bsb[(n4 + 1) * LSTR + kk] = bfbits(wv.y);
            Bsb[(n4 + 2) * LSTR + kk] = bfbits(wv.z);
            Bsb[(n4 + 3) * LSTR + kk] = bfbits(wv.w);
        }
        __syncthreads();

        Frag16 a;   // A layout: row = rloc, K = 8*half + e (+8 for e>=8)
        {
            const unsigned short* ap = Asb + rloc * LSTR + half * 8;
            a.q[0] = *(const uint4*)ap;
            a.q[1] = *(const uint4*)(ap + 16);
        }
        #pragma unroll
        for (int t = 0; t < 4; ++t) {
            Frag16 bfr;  // B layout: col = t*16+l15, K = 8*half + e (+8 for e>=8)
            const unsigned short* bp = Bsb + (t * 16 + l15) * LSTR + half * 8;
            bfr.q[0] = *(const uint4*)bp;
            bfr.q[1] = *(const uint4*)(bp + 16);
            acc[t] = wmma_bf16(a.v, bfr.v, acc[t]);
        }
        __syncthreads();
    }

    // epilogue: bias + scatter to q / k / v-transposed
    #pragma unroll
    for (int t = 0; t < 4; ++t) {
        int cg = colBase + t * 16 + l15;
        float bv = bias[cg];
        #pragma unroll
        for (int r = 0; r < 8; ++r) {
            int row = rowBase + wave * 16 + r + half * 8;
            unsigned short hv = bfbits(acc[t][r] + bv);
            if (cg < 256) {
                qb[row * CDIM + cg] = hv;
            } else if (cg < 512) {
                kb[row * CDIM + (cg - 256)] = hv;
            } else {
                int c = cg - 512, hh = c >> 5, dd = c & 31;
                int b = row >> 12, n = row & 4095;
                vtb[(((b * HEADS + hh) * DHEAD + dd) << 12) + n] = hv;
            }
        }
    }
}

// ---------------------------------------------------------------------------
// Kernel 2: flash attention. Block 256 = 8 waves; wave owns 16 q-rows.
// S^T = K*Q^T (rows=keys, cols=q): softmax stats are in-lane reductions +
// one xor16 shuffle. P rows pack to contiguous bf16 -> b128 LDS stores.
// ---------------------------------------------------------------------------
__global__ __launch_bounds__(256) void flash_kernel(
    const unsigned short* __restrict__ qb, const unsigned short* __restrict__ kb,
    const unsigned short* __restrict__ vtb, unsigned short* __restrict__ ob)
{
    __shared__ unsigned short Pl[8 * 16 * 32];   // wave-private P tiles [q][key]
    const int tid  = threadIdx.x;
    const int lane = tid & 31, wave = tid >> 5;
    const int l15  = lane & 15, half = lane >> 4;
    const int bh = blockIdx.y, b = bh >> 3, h = bh & 7;
    const int qRow = blockIdx.x * 128 + wave * 16;   // [0, 4096)
    const int tokBase = (b << 12);

    // Q^T fragment: B layout, col = q = l15, K = d = 8*half + e (+8)
    Frag16 bq;
    {
        const unsigned short* qp = qb + (size_t)(tokBase + qRow + l15) * CDIM
                                      + h * DHEAD + half * 8;
        bq.q[0] = *(const uint4*)qp;
        bq.q[1] = *(const uint4*)(qp + 16);
    }

    v8f o0{}, o1{};
    float m = -1e30f, lsum = 0.f;
    // fold 1/sqrt(d) and log2(e) into one factor for exp2
    const float kf = 0.17677669529663687f * 1.4426950408889634f;

    unsigned short* Pw = Pl + wave * 512;        // 16x32 bf16 [q][key]
    const unsigned short* vbase = vtb + ((size_t)((b * HEADS + h) * DHEAD) << 12);

    for (int kt = 0; kt < NTOK; kt += 32) {
        // K fragments: A layout, row = key = l15 (+16), K = d = 8*half + e(+8)
        Frag16 ka0, ka1;
        const unsigned short* kp0 = kb + (size_t)(tokBase + kt + l15) * CDIM
                                       + h * DHEAD + half * 8;
        ka0.q[0] = *(const uint4*)kp0;  ka0.q[1] = *(const uint4*)(kp0 + 16);
        const unsigned short* kp1 = kp0 + 16 * CDIM;
        ka1.q[0] = *(const uint4*)kp1;  ka1.q[1] = *(const uint4*)(kp1 + 16);
        __builtin_prefetch(kp0 + 32 * CDIM, 0, 0);

        v8f z{};
        v8f st0 = wmma_bf16(ka0.v, bq.v, z);   // S^T keys kt..kt+15, col q
        v8f st1 = wmma_bf16(ka1.v, bq.v, z);   // S^T keys kt+16..kt+31

        // --- online softmax over keys (in-lane + one xor16) ---
        float tmax = fmaxf(st0[0], st1[0]);
        #pragma unroll
        for (int r = 1; r < 8; ++r)
            tmax = fmaxf(tmax, fmaxf(st0[r], st1[r]));
        tmax = fmaxf(tmax, __shfl_xor(tmax, 16, 32));

        float mnew  = fmaxf(m, tmax);
        float alpha = __builtin_amdgcn_exp2f((m - mnew) * kf);
        m = mnew;
        float mk = mnew * kf;

        float p0[8], p1[8], rsum = 0.f;
        #pragma unroll
        for (int r = 0; r < 8; ++r) {
            p0[r] = __builtin_amdgcn_exp2f(__builtin_fmaf(st0[r], kf, -mk));
            p1[r] = __builtin_amdgcn_exp2f(__builtin_fmaf(st1[r], kf, -mk));
            rsum += p0[r] + p1[r];
        }
        rsum += __shfl_xor(rsum, 16, 32);
        lsum = lsum * alpha + rsum;

        // rescale O rows (alpha indexed by q -> broadcast to O-row owners)
        #pragma unroll
        for (int r = 0; r < 8; ++r) {
            float ar = __shfl(alpha, r + 8 * half, 32);
            o0[r] *= ar;
            o1[r] *= ar;
        }

        // --- P^T lane rows are 8 consecutive keys: pack + b128 store ---
        {
            unsigned short* Pq = Pw + l15 * 32 + half * 8;
            uint4 w0{packbf(p0[0], p0[1]), packbf(p0[2], p0[3]),
                     packbf(p0[4], p0[5]), packbf(p0[6], p0[7])};
            uint4 w1{packbf(p1[0], p1[1]), packbf(p1[2], p1[3]),
                     packbf(p1[4], p1[5]), packbf(p1[6], p1[7])};
            *(uint4*)Pq        = w0;   // keys kt + 8*half + 0..7
            *(uint4*)(Pq + 16) = w1;   // keys kt+16 + 8*half + 0..7
        }
        // P as A fragment: row = q = l15, K = key = 8*half + e (+8)
        Frag16 pa;
        pa.q[0] = *(const uint4*)(Pw + l15 * 32 + half * 8);
        pa.q[1] = *(const uint4*)(Pw + l15 * 32 + half * 8 + 16);

        // V fragments from transposed V: B layout, col = dcol, K = key
        Frag16 vf0, vf1;
        {
            const unsigned short* vp0 = vbase + ((size_t)l15 << 12) + kt + half * 8;
            vf0.q[0] = *(const uint4*)vp0;  vf0.q[1] = *(const uint4*)(vp0 + 16);
            const unsigned short* vp1 = vp0 + (16 << 12);
            vf1.q[0] = *(const uint4*)vp1;  vf1.q[1] = *(const uint4*)(vp1 + 16);
            __builtin_prefetch(vp0 + 32, 0, 0);
        }
        o0 = wmma_bf16(pa.v, vf0.v, o0);
        o1 = wmma_bf16(pa.v, vf1.v, o1);
    }

    // normalize + store attn output bf16 [token][256]
    float linv = __builtin_amdgcn_rcpf(lsum);
    #pragma unroll
    for (int r = 0; r < 8; ++r) {
        float lr = __shfl(linv, r + 8 * half, 32);
        int tok = tokBase + qRow + r + half * 8;
        unsigned short* op = ob + (size_t)tok * CDIM + h * DHEAD;
        op[l15]      = bfbits(o0[r] * lr);
        op[16 + l15] = bfbits(o1[r] * lr);
    }
}

// ---------------------------------------------------------------------------
// Kernel 3: output projection + bias + transpose to [B][C][H*W].
// A (bf16) direct from global; B staged bf16 [col][k] in LDS.
// ---------------------------------------------------------------------------
__global__ __launch_bounds__(128) void proj_kernel(
    const unsigned short* __restrict__ ob, const float* __restrict__ w,
    const float* __restrict__ bias, float* __restrict__ out)
{
    __shared__ unsigned short Bsb[64 * LSTR];   // [col][k] bf16
    const int tid  = threadIdx.x;
    const int lane = tid & 31, wave = tid >> 5;
    const int l15  = lane & 15, half = lane >> 4;
    const int rowBase = blockIdx.x * 64;     // token rows
    const int colBase = blockIdx.y * 64;     // out channels

    v8f acc[4];
    #pragma unroll
    for (int t = 0; t < 4; ++t) acc[t] = v8f{};

    const int rowA = rowBase + wave * 16 + l15;

    for (int k0 = 0; k0 < CDIM; k0 += 32) {
        #pragma unroll
        for (int i = 0; i < 4; ++i) {
            int q  = i * 128 + tid;
            int kk = q >> 4;
            int n4 = (q & 15) * 4;
            float4 wv = *(const float4*)&w[(k0 + kk) * CDIM + colBase + n4];
            Bsb[(n4 + 0) * LSTR + kk] = bfbits(wv.x);
            Bsb[(n4 + 1) * LSTR + kk] = bfbits(wv.y);
            Bsb[(n4 + 2) * LSTR + kk] = bfbits(wv.z);
            Bsb[(n4 + 3) * LSTR + kk] = bfbits(wv.w);
        }
        __syncthreads();

        Frag16 a;   // bf16 A direct from global (contiguous 16B runs)
        {
            const unsigned short* ap = ob + (size_t)rowA * CDIM + k0 + half * 8;
            a.q[0] = *(const uint4*)ap;
            a.q[1] = *(const uint4*)(ap + 16);
        }
        #pragma unroll
        for (int t = 0; t < 4; ++t) {
            Frag16 bfr;
            const unsigned short* bp = Bsb + (t * 16 + l15) * LSTR + half * 8;
            bfr.q[0] = *(const uint4*)bp;
            bfr.q[1] = *(const uint4*)(bp + 16);
            acc[t] = wmma_bf16(a.v, bfr.v, acc[t]);
        }
        __syncthreads();
    }

    #pragma unroll
    for (int t = 0; t < 4; ++t) {
        int cg = colBase + t * 16 + l15;
        float bv = bias[cg];
        #pragma unroll
        for (int r = 0; r < 8; ++r) {
            int row = rowBase + wave * 16 + r + half * 8;
            int bb = row >> 12, n = row & 4095;
            out[(size_t)(bb * CDIM + cg) * NTOK + n] = acc[t][r] + bv;
        }
    }
}

// ---------------------------------------------------------------------------
extern "C" void kernel_launch(void* const* d_in, const int* in_sizes, int n_in,
                              void* d_out, int out_size, void* d_ws, size_t ws_size,
                              hipStream_t stream) {
    const float* x     = (const float*)d_in[0];
    const float* w_qkv = (const float*)d_in[1];
    const float* b_qkv = (const float*)d_in[2];
    const float* w_out = (const float*)d_in[3];
    const float* b_out = (const float*)d_in[4];
    float* out = (float*)d_out;

    // workspace: 4 bf16 buffers of 8192*256 each (16 MiB total)
    unsigned short* qb  = (unsigned short*)d_ws;
    unsigned short* kb  = qb  + (size_t)NROWS * CDIM;
    unsigned short* vtb = kb  + (size_t)NROWS * CDIM;
    unsigned short* ob  = vtb + (size_t)NROWS * CDIM;

    qkv_kernel<<<dim3(NROWS / 64, 768 / 64), 128, 0, stream>>>(
        x, w_qkv, b_qkv, qb, kb, vtb);
    flash_kernel<<<dim3(NTOK / 128, 2 * HEADS), 256, 0, stream>>>(
        qb, kb, vtb, ob);
    proj_kernel<<<dim3(NROWS / 64, CDIM / 64), 128, 0, stream>>>(
        ob, w_out, b_out, out);
}
